// GPTModel_84670985274003
// MI455X (gfx1250) — compile-verified, matrix operand order checked by reading, not run on hardware
//
#include <hip/hip_runtime.h>

// ---------- types ----------
typedef __attribute__((ext_vector_type(16))) __bf16 v16bf;
typedef __attribute__((ext_vector_type(8)))  float  v8f;
typedef __attribute__((ext_vector_type(4)))  unsigned int uint4x;
typedef __attribute__((ext_vector_type(4)))  float  float4x;
typedef unsigned short ushort_t;

// epilogue flags
#define FLAG_BIAS    1
#define FLAG_RELU    2
#define FLAG_RES     4
#define FLAG_BF16OUT 8
#define FLAG_CAUSAL  16
#define FLAG_VTRANS  32
#define FLAG_AF32    64

__device__ __forceinline__ ushort_t f2bf(float x) {
  unsigned u = __float_as_uint(x);
  u += 0x7FFFu + ((u >> 16) & 1u);          // round-to-nearest-even
  return (ushort_t)(u >> 16);
}

union Frag {
  uint4x q[2];
  v16bf  v;
};

// GEMM tile config: 256 threads = 8 waves arranged 4 (M) x 2 (N).
// Workgroup tile 256x128, K-step 32. Wave tile 64x64 = 4x4 wmma 16x16 accums.
// Per K-step per wave: 16 WMMAs vs 8 LDS fragment loads (16 ds_load_b128).
constexpr int BM = 256, BN = 128, BK = 32;
constexpr int LDSP = 40;   // padded row stride (bf16 elems) to spread LDS banks

// C[m,n] = sum_k A[m,k] * B[n,k]   (NT GEMM, both row-major K-contiguous)
template<int FLAGS>
__global__ __launch_bounds__(256)
void gemm_wmma(const void* __restrict__ Av, const ushort_t* __restrict__ Bmat,
               const float* __restrict__ bias, const float* __restrict__ res,
               void* __restrict__ outv,
               int M, int N, int K, int lda, int ldb, int ldc, int Nreal,
               float scale, int Hdiv,
               long long sAo, long long sAi, long long sBo, long long sBi,
               long long sCo, long long sCi)
{
  __shared__ ushort_t Alds[BM * LDSP];   // 256 rows
  __shared__ ushort_t Blds[BN * LDSP];   // 128 rows

  const int tid  = threadIdx.x;
  const int lane = tid & 31;
  const int wave = tid >> 5;
  const int wm   = wave >> 1;          // 0..3  (M direction)
  const int wn   = wave & 1;           // 0..1  (N direction)
  const int lr   = lane & 15;          // row within 16
  const int kg   = (lane >> 4) * 8;    // K sub-offset per half-wave (ISA A/B layout)

  const int tileN = blockIdx.x * BN;
  const int tileM = blockIdx.y * BM;
  const int z     = blockIdx.z;
  const long long offA = (long long)(z / Hdiv) * sAo + (long long)(z % Hdiv) * sAi;
  const long long offB = (long long)(z / Hdiv) * sBo + (long long)(z % Hdiv) * sBi;
  const long long offC = (long long)(z / Hdiv) * sCo + (long long)(z % Hdiv) * sCi;

  const ushort_t* A16 = (const ushort_t*)Av;
  const float*    A32 = (const float*)Av;

  v8f acc[4][4];
  const v8f zero8 = {0.f,0.f,0.f,0.f,0.f,0.f,0.f,0.f};
#pragma unroll
  for (int i = 0; i < 4; ++i)
#pragma unroll
    for (int j = 0; j < 4; ++j) acc[i][j] = zero8;

  for (int kb = 0; kb < K; kb += BK) {
    __syncthreads();
    // ---- global -> LDS ----
    // A: 256x32 = 1024 chunks of 8 bf16, 4 per thread
#pragma unroll
    for (int i = 0; i < 4; ++i) {
      const int c   = tid * 4 + i;        // 0..1023
      const int row = c >> 2;             // 0..255
      const int col = (c & 3) * 8;        // 0,8,16,24
      if (FLAGS & FLAG_AF32) {
        const float* src = A32 + offA + (long long)(tileM + row) * lda + kb + col;
        float4x f0 = *(const float4x*)(src);
        float4x f1 = *(const float4x*)(src + 4);
        uint4x pk;
        pk.x = (unsigned)f2bf(f0.x) | ((unsigned)f2bf(f0.y) << 16);
        pk.y = (unsigned)f2bf(f0.z) | ((unsigned)f2bf(f0.w) << 16);
        pk.z = (unsigned)f2bf(f1.x) | ((unsigned)f2bf(f1.y) << 16);
        pk.w = (unsigned)f2bf(f1.z) | ((unsigned)f2bf(f1.w) << 16);
        *(uint4x*)&Alds[row * LDSP + col] = pk;
      } else {
        uint4x da = *(const uint4x*)(A16 + offA + (long long)(tileM + row) * lda + kb + col);
        *(uint4x*)&Alds[row * LDSP + col] = da;
      }
    }
    // B: 128x32 = 512 chunks of 8 bf16, 2 per thread
#pragma unroll
    for (int i = 0; i < 2; ++i) {
      const int c   = tid * 2 + i;        // 0..511
      const int row = c >> 2;             // 0..127
      const int col = (c & 3) * 8;
      uint4x db = *(const uint4x*)(Bmat + offB + (long long)(tileN + row) * ldb + kb + col);
      *(uint4x*)&Blds[row * LDSP + col] = db;
    }
    __syncthreads();

    // ---- fragments from LDS (per ISA 16-bit 16x32 A/B layout) ----
    Frag a[4], b[4];
#pragma unroll
    for (int mt = 0; mt < 4; ++mt) {
      const int r = wm * 64 + mt * 16 + lr;
      a[mt].q[0] = *(const uint4x*)&Alds[r * LDSP + kg];
      a[mt].q[1] = *(const uint4x*)&Alds[r * LDSP + kg + 16];
    }
#pragma unroll
    for (int nt = 0; nt < 4; ++nt) {
      const int r = wn * 64 + nt * 16 + lr;
      b[nt].q[0] = *(const uint4x*)&Blds[r * LDSP + kg];
      b[nt].q[1] = *(const uint4x*)&Blds[r * LDSP + kg + 16];
    }

#pragma unroll
    for (int mt = 0; mt < 4; ++mt)
#pragma unroll
      for (int nt = 0; nt < 4; ++nt)
        acc[mt][nt] = __builtin_amdgcn_wmma_f32_16x16x32_bf16(
            false, a[mt].v, false, b[nt].v, (short)0, acc[mt][nt], false, false);
  }

  // ---- epilogue ----
  float*    outF = (float*)outv;
  ushort_t* outH = (ushort_t*)outv;
#pragma unroll
  for (int mt = 0; mt < 4; ++mt) {
#pragma unroll
    for (int nt = 0; nt < 4; ++nt) {
      const int n = tileN + wn * 64 + nt * 16 + (lane & 15);
#pragma unroll
      for (int e = 0; e < 8; ++e) {
        const int m = tileM + wm * 64 + mt * 16 + e + (lane >> 4) * 8;
        float vv = acc[mt][nt][e];
        if (FLAGS & FLAG_BIAS)   vv += bias[n];
        if (FLAGS & FLAG_CAUSAL) { vv *= scale; if (n > m) vv = -1e30f; }
        if (FLAGS & FLAG_RELU)   vv = fmaxf(vv, 0.f);
        if (FLAGS & FLAG_VTRANS) {
          // write V transposed into [B,H,128(pad dk),S] bf16
          const int bb = m >> 10, s = m & 1023, hh = n >> 6, d = n & 63;
          const long long oi = (((long long)(bb * 12 + hh) * 128 + d) << 10) + s;
          outH[oi] = f2bf(vv);
        } else if (n < Nreal) {
          const long long oi = offC + (long long)m * ldc + n;
          if (FLAGS & FLAG_RES)     vv += res[oi];
          if (FLAGS & FLAG_BF16OUT) outH[oi] = f2bf(vv);
          else                      outF[oi] = vv;
        }
      }
    }
  }
}

// ---------- LayerNorm: one block per row of 768, writes bf16 ----------
__global__ __launch_bounds__(256)
void layernorm_bf16(const float* __restrict__ x, const float* __restrict__ g,
                    const float* __restrict__ b, ushort_t* __restrict__ y)
{
  const int row = blockIdx.x;
  const float* xr = x + (long long)row * 768;
  __shared__ float red[256];
  const int t = threadIdx.x;
  float xv[3];
  float s = 0.f;
#pragma unroll
  for (int i = 0; i < 3; ++i) { xv[i] = xr[t + i * 256]; s += xv[i]; }
  red[t] = s; __syncthreads();
  for (int w = 128; w > 0; w >>= 1) { if (t < w) red[t] += red[t + w]; __syncthreads(); }
  const float mean = red[0] * (1.f / 768.f);
  __syncthreads();
  float vs = 0.f;
#pragma unroll
  for (int i = 0; i < 3; ++i) { const float d = xv[i] - mean; vs += d * d; }
  red[t] = vs; __syncthreads();
  for (int w = 128; w > 0; w >>= 1) { if (t < w) red[t] += red[t + w]; __syncthreads(); }
  const float rstd = rsqrtf(red[0] * (1.f / 768.f) + 1e-5f);
#pragma unroll
  for (int i = 0; i < 3; ++i) {
    const int c = t + i * 256;
    y[(long long)row * 768 + c] = f2bf((xv[i] - mean) * rstd * g[c] + b[c]);
  }
}

// ---------- causal row softmax over 1024 elems, in-place f32 ----------
__global__ __launch_bounds__(256)
void softmax1024(float* __restrict__ sc)
{
  const long long row = blockIdx.x;
  float* p = sc + (row << 10);
  __shared__ float red[256];
  const int t = threadIdx.x;
  float v[4];
  float mx = -1e38f;
#pragma unroll
  for (int k2 = 0; k2 < 4; ++k2) { v[k2] = p[t + (k2 << 8)]; mx = fmaxf(mx, v[k2]); }
  red[t] = mx; __syncthreads();
  for (int w = 128; w > 0; w >>= 1) { if (t < w) red[t] = fmaxf(red[t], red[t + w]); __syncthreads(); }
  mx = red[0]; __syncthreads();
  float s = 0.f;
#pragma unroll
  for (int k2 = 0; k2 < 4; ++k2) { v[k2] = __expf(v[k2] - mx); s += v[k2]; }
  red[t] = s; __syncthreads();
  for (int w = 128; w > 0; w >>= 1) { if (t < w) red[t] += red[t + w]; __syncthreads(); }
  const float inv = 1.f / red[0];
#pragma unroll
  for (int k2 = 0; k2 < 4; ++k2) p[t + (k2 << 8)] = v[k2] * inv;
}

// ---------- embedding gather: h = tok_emb[x] + pos_emb[:S] ----------
__global__ __launch_bounds__(256)
void embed_kernel(const int* __restrict__ x, const float* __restrict__ te,
                  const float* __restrict__ pe, float* __restrict__ h, int total)
{
  const int i = blockIdx.x * 256 + threadIdx.x;
  if (i >= total) return;
  const int row = i / 768, d = i - row * 768;
  const int tok = x[row];
  h[i] = te[(long long)tok * 768 + d] + pe[(long long)(row & 1023) * 768 + d];
}

// ---------- f32 -> bf16 conversion ----------
__global__ __launch_bounds__(256)
void cvt_bf16(const float* __restrict__ src, ushort_t* __restrict__ dst, long long n)
{
  const long long i = (long long)blockIdx.x * 256 + threadIdx.x;
  if (i < n) dst[i] = f2bf(src[i]);
}

// ---------- host side ----------
template<int FLAGS>
static void launch_gemm(hipStream_t st, const void* A, const ushort_t* B,
                        const float* bias, const float* res, void* out,
                        int M, int N, int K, int lda, int ldb, int ldc, int Nreal,
                        float scale, int Hdiv, int nz,
                        long long sAo, long long sAi, long long sBo, long long sBi,
                        long long sCo, long long sCi)
{
  dim3 grid(N / BN, M / BM, nz);
  gemm_wmma<FLAGS><<<grid, 256, 0, st>>>(A, B, bias, res, out, M, N, K, lda, ldb, ldc,
                                         Nreal, scale, Hdiv, sAo, sAi, sBo, sBi, sCo, sCi);
}

extern "C" void kernel_launch(void* const* d_in, const int* in_sizes, int n_in,
                              void* d_out, int out_size, void* d_ws, size_t ws_size,
                              hipStream_t stream)
{
  (void)in_sizes; (void)n_in; (void)out_size; (void)ws_size;
  constexpr int Bb = 2, S = 1024, D = 768, F = 3072, Hh = 12, L = 4, V = 32000;
  constexpr int M = Bb * S;                     // 2048
  constexpr long long WD = (long long)D * D;    // 589824
  constexpr long long WF = (long long)F * D;    // 2359296

  const int*   x    = (const int*)  d_in[0];
  const float* tok  = (const float*)d_in[1];
  const float* pos  = (const float*)d_in[2];
  const float* Wq   = (const float*)d_in[3];  const float* bq  = (const float*)d_in[4];
  const float* Wk   = (const float*)d_in[5];  const float* bk  = (const float*)d_in[6];
  const float* Wv   = (const float*)d_in[7];  const float* bv  = (const float*)d_in[8];
  const float* Wo   = (const float*)d_in[9];  const float* bo  = (const float*)d_in[10];
  const float* W1   = (const float*)d_in[11]; const float* b1  = (const float*)d_in[12];
  const float* W2   = (const float*)d_in[13]; const float* b2  = (const float*)d_in[14];
  const float* ln1g = (const float*)d_in[15]; const float* ln1b = (const float*)d_in[16];
  const float* ln2g = (const float*)d_in[17]; const float* ln2b = (const float*)d_in[18];
  const float* lnfg = (const float*)d_in[19]; const float* lnfb = (const float*)d_in[20];

  // workspace carve-out
  char* wsp = (char*)d_ws;
  auto carve = [&](size_t bytes) -> char* {
    char* p = wsp; wsp += (bytes + 255) & ~(size_t)255; return p;
  };
  float*    h      = (float*)   carve((size_t)M * D * 4);
  ushort_t* y      = (ushort_t*)carve((size_t)M * D * 2);
  ushort_t* q      = (ushort_t*)carve((size_t)M * D * 2);
  ushort_t* kbuf   = (ushort_t*)carve((size_t)M * D * 2);
  ushort_t* vt     = (ushort_t*)carve((size_t)Bb * Hh * 128 * 1024 * 2); // dk padded to 128
  ushort_t* ctx    = (ushort_t*)carve((size_t)M * D * 2);
  ushort_t* ffn    = (ushort_t*)carve((size_t)M * F * 2);
  float*    scores = (float*)   carve((size_t)Bb * Hh * S * S * 4);
  ushort_t* temb   = (ushort_t*)carve((size_t)V * D * 2);
  ushort_t* wl     = (ushort_t*)carve((size_t)(4 * WD + 2 * WF) * 2);
  ushort_t* wq16 = wl;
  ushort_t* wk16 = wl + WD;
  ushort_t* wv16 = wl + 2 * WD;
  ushort_t* wo16 = wl + 3 * WD;
  ushort_t* w116 = wl + 4 * WD;
  ushort_t* w216 = wl + 4 * WD + WF;

  // embedding + LM-head weight conversion
  {
    const int tot = M * D;
    embed_kernel<<<(tot + 255) / 256, 256, 0, stream>>>(x, tok, pos, h, tot);
    const long long nte = (long long)V * D;
    cvt_bf16<<<(unsigned)((nte + 255) / 256), 256, 0, stream>>>(tok, temb, nte);
  }

  const long long sSD = (long long)S * D;    // per-batch stride of [S,D] tensors
  const long long sSS = (long long)S * S;    // per-head score stride

  for (int l = 0; l < L; ++l) {
    // per-layer weight conversion to bf16
    cvt_bf16<<<(unsigned)((WD + 255) / 256), 256, 0, stream>>>(Wq + l * WD, wq16, WD);
    cvt_bf16<<<(unsigned)((WD + 255) / 256), 256, 0, stream>>>(Wk + l * WD, wk16, WD);
    cvt_bf16<<<(unsigned)((WD + 255) / 256), 256, 0, stream>>>(Wv + l * WD, wv16, WD);
    cvt_bf16<<<(unsigned)((WD + 255) / 256), 256, 0, stream>>>(Wo + l * WD, wo16, WD);
    cvt_bf16<<<(unsigned)((WF + 255) / 256), 256, 0, stream>>>(W1 + l * WF, w116, WF);
    cvt_bf16<<<(unsigned)((WF + 255) / 256), 256, 0, stream>>>(W2 + l * WF, w216, WF);

    // y = LN1(h)  (bf16)
    layernorm_bf16<<<M, 256, 0, stream>>>(h, ln1g + l * D, ln1b + l * D, y);

    // q,k projections (bf16 out); v projection written transposed [B,H,128,S]
    launch_gemm<FLAG_BIAS | FLAG_BF16OUT>(stream, y, wq16, bq + l * D, nullptr, q,
        M, D, D, D, D, D, D, 1.f, 1, 1, 0, 0, 0, 0, 0, 0);
    launch_gemm<FLAG_BIAS | FLAG_BF16OUT>(stream, y, wk16, bk + l * D, nullptr, kbuf,
        M, D, D, D, D, D, D, 1.f, 1, 1, 0, 0, 0, 0, 0, 0);
    launch_gemm<FLAG_BIAS | FLAG_VTRANS>(stream, y, wv16, bv + l * D, nullptr, vt,
        M, D, D, D, D, D, D, 1.f, 1, 1, 0, 0, 0, 0, 0, 0);

    // scores[b,h] = (q_h @ k_h^T) * 1/sqrt(dk), causal mask, f32
    launch_gemm<FLAG_CAUSAL>(stream, q, kbuf, nullptr, nullptr, scores,
        S, S, 64, D, D, S, S, 0.125f, Hh, Bb * Hh,
        sSD, 64, sSD, 64, (long long)Hh * sSS, sSS);

    // softmax rows (in-place f32)
    softmax1024<<<Bb * Hh * S, 256, 0, stream>>>(scores);

    // ctx[b,h] = probs @ v_t^T  (A is f32, converted in-loader; N padded to 128)
    launch_gemm<FLAG_AF32 | FLAG_BF16OUT>(stream, scores, vt, nullptr, nullptr, ctx,
        S, 128, S, S, S, D, 64, 1.f, Hh, Bb * Hh,
        (long long)Hh * sSS, sSS, (long long)Hh * 128 * 1024, 128LL * 1024, sSD, 64);

    // h += ctx @ Wo^T + bo
    launch_gemm<FLAG_BIAS | FLAG_RES>(stream, ctx, wo16, bo + l * D, h, h,
        M, D, D, D, D, D, D, 1.f, 1, 1, 0, 0, 0, 0, 0, 0);

    // y = LN2(h); ffn = relu(y @ W1^T + b1); h += ffn @ W2^T + b2
    layernorm_bf16<<<M, 256, 0, stream>>>(h, ln2g + l * D, ln2b + l * D, y);
    launch_gemm<FLAG_BIAS | FLAG_RELU | FLAG_BF16OUT>(stream, y, w116, b1 + l * F, nullptr, ffn,
        M, F, D, D, D, F, F, 1.f, 1, 1, 0, 0, 0, 0, 0, 0);
    launch_gemm<FLAG_BIAS | FLAG_RES>(stream, ffn, w216, b2 + l * D, h, h,
        M, D, F, F, F, D, D, 1.f, 1, 1, 0, 0, 0, 0, 0, 0);
  }

  // final LN + tied LM head: logits = LNf(h) @ tok_emb^T  (f32 out)
  layernorm_bf16<<<M, 256, 0, stream>>>(h, lnfg, lnfb, y);
  launch_gemm<0>(stream, y, temb, nullptr, nullptr, d_out,
      M, V, D, D, D, V, V, 1.f, 1, 1, 0, 0, 0, 0, 0, 0);
}
